// ScatterCFGEncodedPathsToCFGNodeEncodings_32134945308873
// MI455X (gfx1250) — compile-verified, matrix operand order checked by reading, not run on hardware
//
#include <hip/hip_runtime.h>
#include <hip/hip_bf16.h>
#include <math.h>

#define DDIM 256
#define NEGV (-1e9f)

typedef __attribute__((ext_vector_type(16))) __bf16 v16bf;
typedef __attribute__((ext_vector_type(8)))  float  v8f;

__device__ __forceinline__ unsigned short f2bf(float f) {
    unsigned int b = __float_as_uint(f);
    unsigned int r = b + 0x7FFFu + ((b >> 16) & 1u);   // round-to-nearest-even
    return (unsigned short)(r >> 16);
}

// order-preserving float <-> signed int map (for atomicMax-based segment max)
__device__ __forceinline__ int fmap(float f) {
    int b = __float_as_int(f);
    return (b >= 0) ? b : (b ^ 0x7FFFFFFF);
}
__device__ __forceinline__ float funmap(int m) {
    int b = (m >= 0) ? m : (m ^ 0x7FFFFFFF);
    return __int_as_float(b);
}

// WMMA 16-bit fragment K-index pattern: element i of a lane-half's v16bf
// maps to K = (i<8 ? half*8+i : 16+half*8+(i-8)) within a 32-wide K step.
__device__ __forceinline__ int kidx(int i, int half) {
    return (i < 8) ? (half * 8 + i) : (16 + half * 8 + (i - 8));
}

// ---------------------------------------------------------------------------
// Prep: Af32 = Wq @ Wk^T  (row-major [K=p][N=m], f32 temp)
// ---------------------------------------------------------------------------
__global__ void __launch_bounds__(256) prep_wqwk(const float* __restrict__ Wq,
                                                 const float* __restrict__ Wk,
                                                 float* __restrict__ Af32) {
    __shared__ float wqrow[DDIM];
    int p = blockIdx.x;
    int m = threadIdx.x;
    wqrow[m] = Wq[p * DDIM + m];
    __syncthreads();
    float acc = 0.0f;
#pragma unroll 8
    for (int j = 0; j < DDIM; ++j)
        acc = fmaf(wqrow[j], Wk[m * DDIM + j], acc);
    Af32[p * DDIM + m] = acc;
}

// v[m] = Wk[m,:]·bq ; u[p] = Wq[p,:]·bk ; c0 = bq·bk
__global__ void prep_vec(const float* __restrict__ Wq, const float* __restrict__ Wk,
                         const float* __restrict__ bq, const float* __restrict__ bk,
                         float* __restrict__ vbias, float* __restrict__ uvec,
                         float* __restrict__ c0) {
    int m = threadIdx.x;
    float v = 0.0f, u = 0.0f;
    for (int j = 0; j < DDIM; ++j) {
        v = fmaf(Wk[m * DDIM + j], bq[j], v);
        u = fmaf(Wq[m * DDIM + j], bk[j], u);
    }
    vbias[m] = v;
    uvec[m]  = u;
    if (m == 0) {
        float c = 0.0f;
        for (int j = 0; j < DDIM; ++j) c = fmaf(bq[j], bk[j], c);
        c0[0] = c;
    }
}

// ---------------------------------------------------------------------------
// Swizzle a [K x 256] f32 weight into WMMA B-fragment-major bf16:
//   dst[((ct*KT + kt)*32 + lane)*16 + i] = bf16(W[(kt*32 + kidx(i,half))*256 + ct*16 + lane%16])
// so a lane's whole fragment is 32 contiguous bytes (one b128-pair load).
// ---------------------------------------------------------------------------
__global__ void swizzle_w(const float* __restrict__ W, unsigned short* __restrict__ Wswz,
                          int KT) {
    int total = 16 * KT * 32;           // (ct, kt, lane) triples
    for (int gid = blockIdx.x * blockDim.x + threadIdx.x; gid < total;
         gid += gridDim.x * blockDim.x) {
        int lane = gid & 31;
        int kt   = (gid >> 5) % KT;
        int ct   = gid / (32 * KT);
        int half = lane >> 4;
        int n    = ct * 16 + (lane & 15);
        unsigned short* dst = Wswz + (size_t)gid * 16;
#pragma unroll
        for (int i = 0; i < 16; ++i) {
            int kk = kt * 32 + kidx(i, half);
            dst[i] = f2bf(W[kk * DDIM + n]);
        }
    }
}

// c[i] = prev[i,:]·u + c0
__global__ void cvec_k(const float* __restrict__ prev, const float* __restrict__ uvec,
                       const float* __restrict__ c0, float* __restrict__ cvec, int N) {
    int i = blockIdx.x * blockDim.x + threadIdx.x;
    if (i >= N) return;
    const float* prow = prev + (long long)i * DDIM;
    float acc = c0[0];
#pragma unroll 8
    for (int p = 0; p < DDIM; ++p) acc = fmaf(prow[p], uvec[p], acc);
    cvec[i] = acc;
}

// Map a column index cc (0..255) within a 32-wide K step to (kt, half, i)
__device__ __forceinline__ void amap(int cc, int& kt, int& half, int& i) {
    kt = cc >> 5;
    int ko = cc & 31;
    half = (ko >> 3) & 1;
    i = (ko & 7) + (((ko >> 4) & 1) << 3);
}

// ---------------------------------------------------------------------------
// qk = prev @ A + v   (N x 256 @ 256 x 256, bf16 WMMA, f32 accumulate)
// Block = 256 thr (8 waves); block owns a 16-row tile; wave owns 2 col tiles.
// A-tile staged in LDS already in fragment order -> single 32B ds read/frag.
// ---------------------------------------------------------------------------
__global__ void __launch_bounds__(256) qk_gemm(const float* __restrict__ prev,
                                               const unsigned short* __restrict__ Aswz,
                                               const float* __restrict__ vbias,
                                               float* __restrict__ qk, int N) {
    __shared__ __align__(32) unsigned short sA[8 * 32 * 16];   // 8KB, fragment-major
    int row0 = blockIdx.x * 16;
    for (int t = threadIdx.x; t < 16 * DDIM; t += 256) {
        int r = t >> 8, cc = t & 255;
        int row = row0 + r;
        float pv = (row < N) ? prev[(long long)row * DDIM + cc] : 0.0f;
        int kt, half, i;
        amap(cc, kt, half, i);
        int lane = (half << 4) + r;
        sA[(((kt << 5) + lane) << 4) + i] = f2bf(pv);
    }
    __syncthreads();

    int wave = threadIdx.x >> 5;
    int lane = threadIdx.x & 31;
    int half = lane >> 4;
    const v16bf* sAv = (const v16bf*)sA;
    const v16bf* Wv  = (const v16bf*)Aswz;

#pragma unroll
    for (int tile = 0; tile < 2; ++tile) {
        int ct = wave + tile * 8;
        int n  = ct * 16 + (lane & 15);
        v8f acc = {0.f, 0.f, 0.f, 0.f, 0.f, 0.f, 0.f, 0.f};
#pragma unroll
        for (int kt = 0; kt < 8; ++kt) {
            v16bf av = sAv[kt * 32 + lane];
            v16bf bv = Wv[(ct * 8 + kt) * 32 + lane];
            acc = __builtin_amdgcn_wmma_f32_16x16x32_bf16(
                false, av, false, bv, (short)0, acc, false, false);
        }
        float bias = vbias[n];
        float* orow = qk + (long long)(row0 + half * 8) * DDIM + n;
        if (row0 + 16 <= N) {            // full tile: branch-free stores
#pragma unroll
            for (int r = 0; r < 8; ++r) orow[(long long)r * DDIM] = acc[r] + bias;
        } else {
#pragma unroll
            for (int r = 0; r < 8; ++r)
                if (row0 + half * 8 + r < N) orow[(long long)r * DDIM] = acc[r] + bias;
        }
    }
}

// ---------------------------------------------------------------------------
// Init accumulators (must run every call: atomics below)
// ---------------------------------------------------------------------------
__global__ void init_k(float* __restrict__ combined, float* __restrict__ denom,
                       int* __restrict__ segmax, long long n256, int N) {
    long long stride = (long long)gridDim.x * blockDim.x;
    for (long long gid = (long long)blockIdx.x * blockDim.x + threadIdx.x; gid < n256;
         gid += stride) {
        combined[gid] = 0.0f;
        if (gid < N) {
            denom[gid]  = 0.0f;
            segmax[gid] = INT_MIN;
        }
    }
}

// ---------------------------------------------------------------------------
// Pass 1: score_e = (vals_e · qk[idx_e] + c[idx_e]) / 16 (masked -> NEG);
//         segment max via int-mapped atomicMax.  One wave per edge.
// ---------------------------------------------------------------------------
__global__ void __launch_bounds__(256) score_k(const float* __restrict__ vals,
                                               const int* __restrict__ idx,
                                               const unsigned char* __restrict__ mask,
                                               const float* __restrict__ qk,
                                               const float* __restrict__ cvec,
                                               float* __restrict__ scores,
                                               int* __restrict__ segmax, int E) {
    int wid  = (blockIdx.x * blockDim.x + threadIdx.x) >> 5;
    int lane = threadIdx.x & 31;
    if (wid >= E) return;
    int node = idx[wid];
    const float4* v4 = (const float4*)(vals + (long long)wid * DDIM) + lane * 2;
    const float4* q4 = (const float4*)(qk + (long long)node * DDIM) + lane * 2;
    float4 a0 = v4[0], a1 = v4[1];
    float4 b0 = q4[0], b1 = q4[1];
    float acc = 0.0f;
    acc = fmaf(a0.x, b0.x, acc); acc = fmaf(a0.y, b0.y, acc);
    acc = fmaf(a0.z, b0.z, acc); acc = fmaf(a0.w, b0.w, acc);
    acc = fmaf(a1.x, b1.x, acc); acc = fmaf(a1.y, b1.y, acc);
    acc = fmaf(a1.z, b1.z, acc); acc = fmaf(a1.w, b1.w, acc);
#pragma unroll
    for (int off = 16; off >= 1; off >>= 1) acc += __shfl_xor(acc, off, 32);
    if (lane == 0) {
        float s = (acc + cvec[node]) * 0.0625f;   // 1/sqrt(256)
        s = mask[wid] ? s : NEGV;
        scores[wid] = s;
        atomicMax(segmax + node, fmap(s));
    }
}

// ---------------------------------------------------------------------------
// Pass 2: w = exp(score - segmax) (masked); atomic scatter into denom/combined
// ---------------------------------------------------------------------------
__global__ void __launch_bounds__(256) accum_k(const float* __restrict__ vals,
                                               const int* __restrict__ idx,
                                               const unsigned char* __restrict__ mask,
                                               const float* __restrict__ scores,
                                               const int* __restrict__ segmax,
                                               float* __restrict__ denom,
                                               float* __restrict__ combined, int E) {
    int wid  = (blockIdx.x * blockDim.x + threadIdx.x) >> 5;
    int lane = threadIdx.x & 31;
    if (wid >= E) return;
    if (!mask[wid]) return;                    // e * mask == 0 -> no contribution
    int node = idx[wid];
    int mi   = segmax[node];
    float sm = (mi == INT_MIN) ? 0.0f : funmap(mi);
    float w  = __expf(scores[wid] - sm);
    if (lane == 0) atomicAdd(denom + node, w);
    const float4* v4 = (const float4*)(vals + (long long)wid * DDIM) + lane * 2;
    float4 a0 = v4[0], a1 = v4[1];
    float* crow = combined + (long long)node * DDIM + lane * 8;
    atomicAdd(crow + 0, w * a0.x); atomicAdd(crow + 1, w * a0.y);
    atomicAdd(crow + 2, w * a0.z); atomicAdd(crow + 3, w * a0.w);
    atomicAdd(crow + 4, w * a1.x); atomicAdd(crow + 5, w * a1.y);
    atomicAdd(crow + 6, w * a1.z); atomicAdd(crow + 7, w * a1.w);
}

// ---------------------------------------------------------------------------
// Finalize: normalize combined, h=[prev,combined] staged in LDS fragment-major
// (bf16), two WMMA GEMMs (update & gate, K=512), relu/sigmoid/mix -> out
// ---------------------------------------------------------------------------
__global__ void __launch_bounds__(256) gate_gemm(const float* __restrict__ prev,
                                                 const float* __restrict__ combined,
                                                 const float* __restrict__ denom,
                                                 const unsigned short* __restrict__ WpSwz,
                                                 const unsigned short* __restrict__ WgSwz,
                                                 const float* __restrict__ bp,
                                                 const float* __restrict__ bg,
                                                 float* __restrict__ out, int N) {
    __shared__ __align__(32) unsigned short sH[16 * 32 * 16];  // 16KB, fragment-major
    __shared__ float sP[16][257];                              // prev f32 for epilogue
    int row0 = blockIdx.x * 16;
    for (int t = threadIdx.x; t < 2 * 16 * DDIM; t += 256) {
        int srcsel = t >> 12;                // 0: prev, 1: normalized combined
        int tt = t & 4095;
        int r = tt >> 8, cc = tt & 255;
        int row = row0 + r;
        float v = 0.0f;
        if (row < N) {
            if (srcsel == 0) {
                v = prev[(long long)row * DDIM + cc];
                sP[r][cc] = v;
            } else {
                v = combined[(long long)row * DDIM + cc] / fmaxf(denom[row], 1e-9f);
            }
        } else if (srcsel == 0) {
            sP[r][cc] = 0.0f;
        }
        int kt, half, i;
        amap(cc, kt, half, i);
        kt += srcsel * 8;                    // K 0..255 = prev, 256..511 = combined
        int lane = (half << 4) + r;
        sH[(((kt << 5) + lane) << 4) + i] = f2bf(v);
    }
    __syncthreads();

    int wave = threadIdx.x >> 5;
    int lane = threadIdx.x & 31;
    int half = lane >> 4;
    const v16bf* sHv = (const v16bf*)sH;
    const v16bf* Wpv = (const v16bf*)WpSwz;
    const v16bf* Wgv = (const v16bf*)WgSwz;

#pragma unroll
    for (int tile = 0; tile < 2; ++tile) {
        int ct = wave + tile * 8;
        int n  = ct * 16 + (lane & 15);
        __builtin_prefetch(WpSwz + (size_t)(ct * 16) * 32 * 16, 0, 3);
        __builtin_prefetch(WgSwz + (size_t)(ct * 16) * 32 * 16, 0, 3);
        v8f ap = {0.f, 0.f, 0.f, 0.f, 0.f, 0.f, 0.f, 0.f};
        v8f ag = {0.f, 0.f, 0.f, 0.f, 0.f, 0.f, 0.f, 0.f};
#pragma unroll
        for (int kt = 0; kt < 16; ++kt) {
            v16bf av  = sHv[kt * 32 + lane];
            v16bf bpv_ = Wpv[(ct * 16 + kt) * 32 + lane];
            v16bf bgv_ = Wgv[(ct * 16 + kt) * 32 + lane];
            ap = __builtin_amdgcn_wmma_f32_16x16x32_bf16(
                false, av, false, bpv_, (short)0, ap, false, false);
            ag = __builtin_amdgcn_wmma_f32_16x16x32_bf16(
                false, av, false, bgv_, (short)0, ag, false, false);
        }
        float bpv = bp[n], bgv = bg[n];
        float* orow = out + (long long)(row0 + half * 8) * DDIM + n;
        if (row0 + 16 <= N) {               // full tile: branch-free epilogue
#pragma unroll
            for (int r = 0; r < 8; ++r) {
                float up = fmaxf(ap[r] + bpv, 0.0f);
                float g  = 1.0f / (1.0f + __expf(-(ag[r] + bgv)));
                float pv = sP[r + half * 8][n];
                orow[(long long)r * DDIM] = g * pv + (1.0f - g) * up;
            }
        } else {
#pragma unroll
            for (int r = 0; r < 8; ++r) {
                if (row0 + half * 8 + r < N) {
                    float up = fmaxf(ap[r] + bpv, 0.0f);
                    float g  = 1.0f / (1.0f + __expf(-(ag[r] + bgv)));
                    float pv = sP[r + half * 8][n];
                    orow[(long long)r * DDIM] = g * pv + (1.0f - g) * up;
                }
            }
        }
    }
}

// ---------------------------------------------------------------------------
extern "C" void kernel_launch(void* const* d_in, const int* in_sizes, int n_in,
                              void* d_out, int out_size, void* d_ws, size_t ws_size,
                              hipStream_t stream) {
    const float*         enc   = (const float*)d_in[0];
    const unsigned char* maskp = (const unsigned char*)d_in[1];
    const int*           idxp  = (const int*)d_in[2];
    const float*         prev  = (const float*)d_in[3];
    // d_in[4] = nr_cfg_nodes (device scalar) -- N derived from sizes instead
    const float* Wq = (const float*)d_in[5];
    const float* bq = (const float*)d_in[6];
    const float* Wk = (const float*)d_in[7];
    const float* bk = (const float*)d_in[8];
    const float* Wp = (const float*)d_in[9];
    const float* bp = (const float*)d_in[10];
    const float* Wg = (const float*)d_in[11];
    const float* bg = (const float*)d_in[12];
    float* out = (float*)d_out;

    const int E = in_sizes[2];            // 262144 edges
    const int N = in_sizes[3] / DDIM;     // 50000 nodes

    // Workspace carve-out
    char* w = (char*)d_ws;
    auto carve = [&](size_t bytes) {
        char* p = w;
        w += (bytes + 255) & ~(size_t)255;
        return p;
    };
    float*          Af32  = (float*)carve((size_t)DDIM * DDIM * 4);
    unsigned short* Aswz  = (unsigned short*)carve((size_t)DDIM * DDIM * 2);
    float*          vbias = (float*)carve(DDIM * 4);
    float*          uvec  = (float*)carve(DDIM * 4);
    float*          c0    = (float*)carve(256);
    unsigned short* WpSwz = (unsigned short*)carve((size_t)2 * DDIM * DDIM * 2);
    unsigned short* WgSwz = (unsigned short*)carve((size_t)2 * DDIM * DDIM * 2);
    float*          qk    = (float*)carve((size_t)N * DDIM * 4);
    float*          cvec  = (float*)carve((size_t)N * 4);
    float*          scr   = (float*)carve((size_t)E * 4);
    int*            smax  = (int*)carve((size_t)N * 4);
    float*          den   = (float*)carve((size_t)N * 4);
    float*          comb  = (float*)carve((size_t)N * DDIM * 4);

    prep_wqwk<<<DDIM, DDIM, 0, stream>>>(Wq, Wk, Af32);
    prep_vec<<<1, DDIM, 0, stream>>>(Wq, Wk, bq, bk, vbias, uvec, c0);
    swizzle_w<<<16, 256, 0, stream>>>(Af32, Aswz, 8);      // A: K=256 -> 8 k-steps
    swizzle_w<<<32, 256, 0, stream>>>(Wp, WpSwz, 16);      // Wp: K=512 -> 16 k-steps
    swizzle_w<<<32, 256, 0, stream>>>(Wg, WgSwz, 16);      // Wg: K=512 -> 16 k-steps
    cvec_k<<<(N + 255) / 256, 256, 0, stream>>>(prev, uvec, c0, cvec, N);
    qk_gemm<<<(N + 15) / 16, 256, 0, stream>>>(prev, Aswz, vbias, qk, N);
    init_k<<<4096, 256, 0, stream>>>(comb, den, smax, (long long)N * DDIM, N);
    score_k<<<(E + 7) / 8, 256, 0, stream>>>(enc, idxp, maskp, qk, cvec, scr, smax, E);
    accum_k<<<(E + 7) / 8, 256, 0, stream>>>(enc, idxp, maskp, scr, smax, den, comb, E);
    gate_gemm<<<(N + 15) / 16, 256, 0, stream>>>(prev, comb, den, WpSwz, WgSwz, bp, bg, out, N);
}
